// Model_21998822490499
// MI455X (gfx1250) — matrix operand order repaired and checked by reading, not compiled, not run on hardware
//
#include <hip/hip_runtime.h>
#include <hip/hip_bf16.h>
#include <math.h>

// ---------------------------------------------------------------------------
// BRITS-style bidirectional LSTM imputation model for MI455X (gfx1250).
// Matrix work done with v_wmma_f32_16x16x32_f16 (f16 in, f32 accumulate).
// One wave32 owns 16 batch rows and carries (h,c) through all T=48 steps.
// A-operands live in LDS (two ds_load_b128 per fragment); weights stream from
// L2 as fragment-contiguous f16 (two global b128 per fragment, re-read each
// step on purpose -- the pointer is laundered through an empty asm so LICM
// cannot hoist 8KB of weights out of the time loop and spill them to scratch,
// then cast back to address_space(1) so the loads stay on the GLOBAL path
// (LOADcnt only) instead of FLAT (LOADcnt+DScnt, which would couple weight
// waits to the LDS h traffic).
// ---------------------------------------------------------------------------

typedef __attribute__((ext_vector_type(16))) _Float16 v16h;
typedef __attribute__((ext_vector_type(8)))  _Float16 v8h;
typedef __attribute__((ext_vector_type(8)))  float    v8f;

typedef const v16h __attribute__((address_space(1)))* wgptr;  // global-AS weights

constexpr int Bn = 8192;
constexpr int Tn = 48;
constexpr int Fn = 35;
constexpr int Hn = 128;          // hidden
constexpr int Gn = 4 * Hn;       // 512 gate width
constexpr int NTILES = Gn / 16;  // 32 N-tiles of the gate GEMM

__device__ __forceinline__ float sigf(float x) { return 1.0f / (1.0f + expf(-x)); }
// fast device transcendentals (v_exp_f32 / v_rcp_f32)
__device__ __forceinline__ float fsig(float x) {
    return __builtin_amdgcn_rcpf(1.0f + __expf(-x));
}
__device__ __forceinline__ float ftanh(float x) {
    // 1 - 2/(exp(2x)+1): saturates to +-1 correctly
    return 1.0f - 2.0f * __builtin_amdgcn_rcpf(__expf(2.0f * x) + 1.0f);
}

// ---------------------------------------------------------------------------
// Pack W_ih^T (K x 512) on top of W_hh^T (128 x 512), zero-padding K_x up to
// x_slabs*32, into B-fragment-contiguous f16:
//   dst[((slab*32 + ntile)*32 + lane)*16 + hh]
// with B-frag mapping  K = slab*32 + (hh&7) + (lane>>4)*8 + (hh>>3)*16 , N = ntile*16 + (lane&15)
// ---------------------------------------------------------------------------
__global__ void pack_w_kernel(const float* __restrict__ Wih,
                              const float* __restrict__ Whh,
                              _Float16* __restrict__ dst,
                              int Kin, int x_slabs, int slabs) {
    int i = blockIdx.x * 256 + threadIdx.x;
    int total = slabs * 32 * 32 * 16;
    if (i >= total) return;
    int hh    = i & 15;
    int lane  = (i >> 4) & 31;
    int ntile = (i >> 9) & 31;
    int slab  = i >> 14;
    int k = slab * 32 + (hh & 7) + ((lane >> 4) << 3) + ((hh >> 3) << 4);
    int n = ntile * 16 + (lane & 15);
    int kxb = x_slabs * 32;
    float v = 0.0f;
    if (k < Kin)        v = Wih[(size_t)n * Kin + k];
    else if (k >= kxb)  v = Whh[(size_t)n * Hn + (k - kxb)];
    dst[i] = (_Float16)v;
}

__global__ void add2_kernel(const float* a, const float* b, float* o, int n) {
    int i = blockIdx.x * 256 + threadIdx.x;
    if (i < n) o[i] = a[i] + b[i];
}

__global__ void zero_kernel(float* p, int n) {
    int i = blockIdx.x * 128 + threadIdx.x;
    if (i < n) p[i] = 0.0f;
}

// ---------------------------------------------------------------------------
// Persistent LSTM scan. One wave = 16 batch rows, full T steps.
// XS = number of 32-wide K-slabs for the input part (4 -> Kx pad of 105,
// 2 -> Kx pad of 35). CONCAT3: input row = [values|masks|deltas].
// The per-row A operand lives in LDS: cols [0, XS*32) = x (f16), cols
// [XS*32, XS*32+128) = h (f16), zero-padded; h region doubles as the carry.
// h_out (f16 [T][B][H]) gets the PRE-update hidden; hT_out (f16 [B][H]) the
// final hidden. reverse_store writes scan step s at t = T-1-s.
// ---------------------------------------------------------------------------
template <int XS, bool CONCAT3>
__global__ __launch_bounds__(128, 1)
void lstm_kernel(const float* __restrict__ in0,
                 const float* __restrict__ in1,
                 const float* __restrict__ in2,
                 const _Float16* __restrict__ wpack,  // [(XS+4)][32][32][16] f16
                 const float* __restrict__ bias,      // [512]
                 _Float16* __restrict__ h_out,
                 _Float16* __restrict__ hT_out,
                 int reverse_store) {
    constexpr int SLABS = XS + 4;
    constexpr int KX  = XS * 32;        // x columns
    constexpr int ROW = KX + Hn + 8;    // padded LDS row (halves), 16B-aligned
    const int lane = threadIdx.x & 31;
    const int wv   = threadIdx.x >> 5;
    const int mrow = lane & 15;         // A row / B,C column within tile
    const int khi  = lane >> 4;         // half-select
    const int base = (blockIdx.x * 4 + wv) * 16;   // first batch row of wave

    __shared__ _Float16 ash_all[4][16][ROW];
    __shared__ float bsh[Gn];
    _Float16 (*ash)[ROW] = ash_all[wv];

    for (int i = threadIdx.x; i < Gn; i += 128) bsh[i] = bias[i];
    {   // zero this wave's A buffer (x pad cols stay zero forever; h starts 0)
        _Float16* p = &ash[0][0];
        for (int i = lane; i < 16 * ROW; i += 32) p[i] = (_Float16)0.0f;
    }
    __syncthreads();

    const v8f vz = {0.f, 0.f, 0.f, 0.f, 0.f, 0.f, 0.f, 0.f};
    v8f c[8];
#pragma unroll
    for (int i = 0; i < 8; ++i) c[i] = vz;

    const int cprow = lane >> 1;          // copy row for h emit
    const int cpc0  = (lane & 1) * 64;    // copy col base

#pragma unroll 1
    for (int t = 0; t < Tn; ++t) {
        // ---- emit pre-update hidden: LDS h region -> h_out (f16) ----
        if (h_out) {
            int ts = reverse_store ? (Tn - 1 - t) : t;
            const v8h* src = (const v8h*)&ash[cprow][KX + cpc0];
            v8h* dst = (v8h*)&h_out[((size_t)ts * Bn + base + cprow) * Hn + cpc0];
#pragma unroll
            for (int q = 0; q < 4; ++q) dst[q] = src[q];
        }

        // ---- stage input rows into LDS (f32 -> f16), coalesced 32-bit offs --
        {
            constexpr int XTOT = (CONCAT3 ? 3 : 1) * Fn * 16;
#pragma unroll 1
            for (int j = lane; j < XTOT; j += 32) {
                int part, rem;
                if (CONCAT3) { part = j / (Fn * 16); rem = j - part * (Fn * 16); }
                else         { part = 0;             rem = j; }
                int row = rem / Fn;
                int kk  = rem - row * Fn;
                const float* src = CONCAT3 ? (part == 0 ? in0 : (part == 1 ? in1 : in2))
                                           : in0;
                unsigned off = (unsigned)(((base + row) * Tn + t) * Fn + kk);
                ash[row][part * Fn + kk] = (_Float16)src[off];
            }
        }

        // ---- build A fragments: two aligned ds_load_b128 per slab ----
        v16h a[SLABS];
#pragma unroll
        for (int s = 0; s < SLABS; ++s) {
            v8h lo = *(const v8h*)&ash[mrow][s * 32 + khi * 8];
            v8h hi = *(const v8h*)&ash[mrow][s * 32 + 16 + khi * 8];
            a[s] = __builtin_shufflevector(lo, hi,
                    0, 1, 2, 3, 4, 5, 6, 7, 8, 9, 10, 11, 12, 13, 14, 15);
        }

        // ---- gate GEMM + elementwise per H-tile ----
#pragma unroll
        for (int hi = 0; hi < 8; ++hi) {
            // Launder the weight base pointer so LICM cannot hoist the (t-
            // invariant) fragment loads out of the time loop and spill 8KB to
            // scratch; also bounds scheduler look-ahead to one 4-gate group.
            // Cast back to address_space(1) so loads lower as global_load
            // (LOADcnt only), not flat_load (LOADcnt+DScnt).
            uintptr_t wq = (uintptr_t)wpack;
            asm volatile("" : "+s"(wq));
            wgptr wpt = (wgptr)wq;

            v8f acc[4];
#pragma unroll
            for (int g = 0; g < 4; ++g) {   // torch order i,f,g,o
                int ntile = g * 8 + hi;
                v8f r = vz;
#pragma unroll
                for (int s = 0; s < SLABS; ++s) {
                    v16h bf = wpt[(s * NTILES + ntile) * 32 + lane];
                    r = __builtin_amdgcn_wmma_f32_16x16x32_f16(
                            false, a[s], false, bf, (short)0, r, false, false);
                }
                float bn = bsh[ntile * 16 + mrow];
#pragma unroll
                for (int e = 0; e < 8; ++e) r[e] += bn;
                acc[g] = r;
            }
#pragma unroll
            for (int e = 0; e < 8; ++e) {
                float ig = fsig(acc[0][e]);
                float fg = fsig(acc[1][e]);
                float gg = ftanh(acc[2][e]);
                float og = fsig(acc[3][e]);
                float c2 = fg * c[hi][e] + ig * gg;
                c[hi][e] = c2;
                float h2 = og * ftanh(c2);
                ash[e + khi * 8][KX + hi * 16 + mrow] = (_Float16)h2;
            }
        }
    }

    if (hT_out) {
        const v8h* src = (const v8h*)&ash[cprow][KX + cpc0];
        v8h* dst = (v8h*)&hT_out[((size_t)(base + cprow)) * Hn + cpc0];
#pragma unroll
        for (int q = 0; q < 4; ++q) dst[q] = src[q];
    }
}

// ---------------------------------------------------------------------------
// Imputation head: x_h = [hf|hb] @ hist_W^T + hist_b ; alpha = sig(md@wc^T+b);
// x_c = alpha*x_h + (1-alpha). Writes imputations [B][T][F] and accumulates
// per-t |x - x_c|*m and m. Block = 8 batch rows of one t.
// ---------------------------------------------------------------------------
__global__ __launch_bounds__(288)
void impute_kernel(const _Float16* __restrict__ hf, const _Float16* __restrict__ hb,
                   const float* __restrict__ values_f, const float* __restrict__ masks_f,
                   const float* __restrict__ deltas_f,
                   const float* __restrict__ hist_W, const float* __restrict__ hist_b,
                   const float* __restrict__ wc_W, const float* __restrict__ wc_b,
                   float* __restrict__ imput_out,
                   float* __restrict__ num_acc, float* __restrict__ den_acc) {
    const int t   = blockIdx.y;
    const int b0  = blockIdx.x * 8;
    const int tid = threadIdx.x;

    __shared__ float hc[8][2 * Hn];   // 8 x 256
    __shared__ float md[8][2 * Fn];   // 8 x 70
    __shared__ float rn[288], rd[288];

    for (int i = tid; i < 8 * 2 * Hn; i += 288) {
        int r = i >> 8, j = i & 255;
        float v = (j < Hn) ? (float)hf[((size_t)t * Bn + b0 + r) * Hn + j]
                           : (float)hb[((size_t)t * Bn + b0 + r) * Hn + (j - Hn)];
        hc[r][j] = v;
    }
    for (int i = tid; i < 8 * 2 * Fn; i += 288) {
        int r = i / (2 * Fn), j = i % (2 * Fn);
        float v = (j < Fn) ? masks_f[((size_t)(b0 + r) * Tn + t) * Fn + j]
                           : deltas_f[((size_t)(b0 + r) * Tn + t) * Fn + (j - Fn)];
        md[r][j] = v;
    }
    __syncthreads();

    float lnum = 0.0f, lden = 0.0f;
    if (tid < 8 * Fn) {
        int r = tid / Fn, f = tid % Fn;
        int b = b0 + r;
        float xh = hist_b[f];
#pragma unroll 8
        for (int j = 0; j < 2 * Hn; ++j) xh += hc[r][j] * hist_W[f * 2 * Hn + j];
        float aa = wc_b[f];
#pragma unroll 5
        for (int j = 0; j < 2 * Fn; ++j) aa += md[r][j] * wc_W[f * 2 * Fn + j];
        float alpha = sigf(aa);
        float xc = alpha * xh + (1.0f - alpha);
        imput_out[((size_t)b * Tn + t) * Fn + f] = xc;
        float m  = masks_f[((size_t)b * Tn + t) * Fn + f];
        float xt = values_f[((size_t)b * Tn + t) * Fn + f];
        lnum = fabsf(xt - xc) * m;
        lden = m;
    }
    rn[tid] = lnum; rd[tid] = lden;
    __syncthreads();
    for (int s = 256; s > 0; s >>= 1) {
        if (tid < s && tid + s < 288) { rn[tid] += rn[tid + s]; rd[tid] += rd[tid + s]; }
        __syncthreads();
    }
    if (tid == 0) {
        atomicAdd(&num_acc[t], rn[0]);
        atomicAdd(&den_acc[t], rd[0]);
    }
}

// ---------------------------------------------------------------------------
// Output head: y = hT @ out_W^T + out_b, predictions, masked BCE sums.
// ---------------------------------------------------------------------------
__global__ __launch_bounds__(256)
void head_kernel(const _Float16* __restrict__ hT, const float* __restrict__ out_W,
                 const float* __restrict__ out_b, const float* __restrict__ labels,
                 const float* __restrict__ is_train,
                 float* __restrict__ pred_out,
                 float* __restrict__ ysum, float* __restrict__ tsum) {
    __shared__ float ry[256], rt[256];
    const int tid = threadIdx.x;
    const int b = blockIdx.x * 256 + tid;
    float lb = 0.0f, lt = 0.0f;
    if (b < Bn) {
        float y = out_b[0];
#pragma unroll 8
        for (int j = 0; j < Hn; ++j) y += (float)hT[(size_t)b * Hn + j] * out_W[j];
        pred_out[b] = sigf(y);
        float lab = labels[b];
        float it  = is_train[b];
        float mv  = fmaxf(-y, 0.0f);
        float bce = y - y * lab + mv + logf(expf(-mv) + expf(-y - mv));
        lb = bce * it;
        lt = it;
    }
    ry[tid] = lb; rt[tid] = lt;
    __syncthreads();
    for (int s = 128; s > 0; s >>= 1) {
        if (tid < s) { ry[tid] += ry[tid + s]; rt[tid] += rt[tid + s]; }
        __syncthreads();
    }
    if (tid == 0) {
        atomicAdd(ysum, ry[0]);
        atomicAdd(tsum, rt[0]);
    }
}

__global__ void finalize_kernel(const float* num, const float* den,
                                const float* ysum, const float* tsum,
                                float* loss_out) {
    if (threadIdx.x == 0 && blockIdx.x == 0) {
        float xl = 0.0f;
        for (int t = 0; t < Tn; ++t) xl += num[t] / (den[t] + 1e-5f);
        float yl = ysum[0] / (tsum[0] + 1e-5f);
        loss_out[0] = 0.3f * xl + 1.0f * yl;
    }
}

// ---------------------------------------------------------------------------
extern "C" void kernel_launch(void* const* d_in, const int* in_sizes, int n_in,
                              void* d_out, int out_size, void* d_ws, size_t ws_size,
                              hipStream_t stream) {
    (void)in_sizes; (void)n_in; (void)out_size; (void)ws_size;
    const float* values_f = (const float*)d_in[0];
    const float* masks_f  = (const float*)d_in[1];
    const float* deltas_f = (const float*)d_in[2];
    const float* values_b = (const float*)d_in[3];
    const float* masks_b  = (const float*)d_in[4];
    const float* deltas_b = (const float*)d_in[5];
    const float* labels   = (const float*)d_in[6];
    const float* is_train = (const float*)d_in[7];
    const float* W_ih  = (const float*)d_in[8];
    const float* W_hh  = (const float*)d_in[9];
    const float* b_ih  = (const float*)d_in[10];
    const float* b_hh  = (const float*)d_in[11];
    const float* hist_W = (const float*)d_in[12];
    const float* hist_b = (const float*)d_in[13];
    // d_in[14], d_in[15]: feat_W/feat_b unused (repo bug preserved by reference)
    const float* wc_W  = (const float*)d_in[16];
    const float* wc_b  = (const float*)d_in[17];
    const float* pW_ih = (const float*)d_in[18];
    const float* pW_hh = (const float*)d_in[19];
    const float* pb_ih = (const float*)d_in[20];
    const float* pb_hh = (const float*)d_in[21];
    const float* out_W = (const float*)d_in[22];
    const float* out_b = (const float*)d_in[23];

    // workspace carve-out
    char* ws = (char*)d_ws;
    size_t off = 0;
    auto carve = [&](size_t bytes) {
        void* p = ws + off;
        off = (off + bytes + 255) & ~(size_t)255;
        return p;
    };
    _Float16* wimp  = (_Float16*)carve((size_t)8 * 32 * 32 * 16 * 2);  // 256 KB
    _Float16* wpred = (_Float16*)carve((size_t)6 * 32 * 32 * 16 * 2);  // 192 KB
    float* biasc = (float*)carve(Gn * 4);
    float* pbias = (float*)carve(Gn * 4);
    _Float16* hf = (_Float16*)carve((size_t)Tn * Bn * Hn * 2);  // ~100 MB
    _Float16* hb = (_Float16*)carve((size_t)Tn * Bn * Hn * 2);  // ~100 MB
    _Float16* hT = (_Float16*)carve((size_t)Bn * Hn * 2);       // 2 MB
    float* accum = (float*)carve(128 * 4);
    float* numA = accum;        // [48]
    float* denA = accum + 48;   // [48]
    float* ysum = accum + 96;
    float* tsum = accum + 97;

    float* loss_out  = (float*)d_out;
    float* pred_out  = loss_out + 1;
    float* imput_out = pred_out + Bn;   // [B][T][F]

    // 0) zero accumulators (must be re-done every call)
    zero_kernel<<<1, 128, 0, stream>>>(accum, 128);

    // 1) weight packing + combined biases
    pack_w_kernel<<<(8 * 32 * 32 * 16 + 255) / 256, 256, 0, stream>>>(
        W_ih, W_hh, wimp, 3 * Fn /*105*/, 4, 8);
    pack_w_kernel<<<(6 * 32 * 32 * 16 + 255) / 256, 256, 0, stream>>>(
        pW_ih, pW_hh, wpred, Fn /*35*/, 2, 6);
    add2_kernel<<<2, 256, 0, stream>>>(b_ih, b_hh, biasc, Gn);
    add2_kernel<<<2, 256, 0, stream>>>(pb_ih, pb_hh, pbias, Gn);

    // 2) forward + backward imputation LSTM scans (pre-update h emitted)
    dim3 lgrid(Bn / (16 * 4));  // 128 blocks x 4 waves x 16 rows = 8192
    lstm_kernel<4, true><<<lgrid, 128, 0, stream>>>(
        values_f, masks_f, deltas_f, wimp, biasc, hf, nullptr, 0);
    lstm_kernel<4, true><<<lgrid, 128, 0, stream>>>(
        values_b, masks_b, deltas_b, wimp, biasc, hb, nullptr, 1);

    // 3) imputation head + per-t loss accumulation
    impute_kernel<<<dim3(Bn / 8, Tn), 288, 0, stream>>>(
        hf, hb, values_f, masks_f, deltas_f, hist_W, hist_b, wc_W, wc_b,
        imput_out, numA, denA);

    // 4) prediction LSTM over imputations (final hidden only)
    lstm_kernel<2, false><<<lgrid, 128, 0, stream>>>(
        imput_out, nullptr, nullptr, wpred, pbias, nullptr, hT, 0);

    // 5) output head + loss finalize
    head_kernel<<<Bn / 256, 256, 0, stream>>>(
        hT, out_W, out_b, labels, is_train, pred_out, ysum, tsum);
    finalize_kernel<<<1, 1, 0, stream>>>(numA, denA, ysum, tsum, loss_out);
}
